// MultiviewEncoder_2216203125207
// MI455X (gfx1250) — matrix operand order repaired and checked by reading, not compiled, not run on hardware
//
#include <hip/hip_runtime.h>
#include <math.h>

// ---------------------------------------------------------------------------
// MultiviewEncoder for MI455X (gfx1250): bf16 WMMA everywhere, fused
// flash-style attention strip in 268KB of WGP LDS, async global->LDS staging,
// nontemporal streaming for the 256MB attn output / 64MB adjacency read.
// sched_barrier keeps fragment-load batches ahead of WMMA batches so the
// compiler emits partial s_wait_loadcnt instead of full drains.
// ---------------------------------------------------------------------------

#define NTOK 4096

typedef __attribute__((ext_vector_type(16))) __bf16 v16bf;
typedef __attribute__((ext_vector_type(8)))  float  v8f;
typedef float fv4 __attribute__((ext_vector_type(4)));
typedef unsigned int uv4 __attribute__((ext_vector_type(4)));

union Frag16 {
  v16bf v;
  uv4 q[2];
  unsigned short u[16];
};

#if __has_builtin(__builtin_amdgcn_sched_barrier)
#define SCHED_FENCE() __builtin_amdgcn_sched_barrier(0)
#else
#define SCHED_FENCE()
#endif

__device__ __forceinline__ unsigned short f32_to_bf16(float f) {
  unsigned u = __float_as_uint(f);
  u += 0x7FFFu + ((u >> 16) & 1u);   // round-to-nearest-even
  return (unsigned short)(u >> 16);
}

__device__ __forceinline__ v8f vzero() {
  v8f z;
#pragma unroll
  for (int i = 0; i < 8; ++i) z[i] = 0.0f;
  return z;
}

// A fragment: 16x32 bf16, M = lane&15. lanes 0-15 hold K {0..7,16..23},
// lanes 16-31 hold K {8..15,24..31}  (ISA 7.12.2, 16-bit A 16x32).
__device__ __forceinline__ void load_a_frag(Frag16& f, const unsigned short* A,
                                            int lda, int row0, int k0, int lane) {
  const unsigned short* p =
      A + (size_t)(row0 + (lane & 15)) * lda + k0 + ((lane >> 4) << 3);
  f.q[0] = *(const uv4*)p;
  f.q[1] = *(const uv4*)(p + 16);
}

// B fragment from a transposed operand Bt[N][K]: lane = output column n,
// lanes 0-15 hold K {0..7,16..23}, lanes 16-31 hold K {8..15,24..31}.
__device__ __forceinline__ void load_b_frag(Frag16& f, const unsigned short* Bt,
                                            int ldb, int col0, int k0, int lane) {
  const unsigned short* p =
      Bt + (size_t)(col0 + (lane & 15)) * ldb + k0 + ((lane >> 4) << 3);
  f.q[0] = *(const uv4*)p;
  f.q[1] = *(const uv4*)(p + 16);
}

__device__ __forceinline__ v8f wmma_bf16(const Frag16& a, const Frag16& b, v8f c) {
  return __builtin_amdgcn_wmma_f32_16x16x32_bf16(false, a.v, false, b.v,
                                                 (short)0, c, false, false);
}

__device__ __forceinline__ float gelu_exact(float x) {
  return 0.5f * x * (1.0f + erff(x * 0.7071067811865476f));
}

// ---------------------------------------------------------------------------
// Kernel 1: build bf16 activations: Z16[N,256], Zg16/Zc16 = concat(x, PE)[N,128]
// ---------------------------------------------------------------------------
__global__ __launch_bounds__(256) void prep_kernel(
    const float* __restrict__ Z, const float* __restrict__ Zg,
    const float* __restrict__ Zc, unsigned short* __restrict__ Z16,
    unsigned short* __restrict__ Zg16, unsigned short* __restrict__ Zc16) {
  int i = blockIdx.x;
  int t = threadIdx.x;
  Z16[(size_t)i * 256 + t] = f32_to_bf16(Z[(size_t)i * 256 + t]);
  if (t < 128) {
    float pe = 0.0f;
    if (t >= 64) {
      int d = t - 64;
      int de = d & ~1;
      float div = __expf((float)de * (-9.210340371976184f / 64.0f)); // -ln(1e4)/64
      float ang = (float)i * div;
      pe = (d & 1) ? __cosf(ang) : __sinf(ang);
    }
    float zg = (t < 64) ? Zg[(size_t)i * 64 + t] : pe;
    float zc = (t < 64) ? Zc[(size_t)i * 64 + t] : pe;
    Zg16[(size_t)i * 128 + t] = f32_to_bf16(zg);
    Zc16[(size_t)i * 128 + t] = f32_to_bf16(zc);
  }
}

// ---------------------------------------------------------------------------
// Kernel 2: convert+transpose a weight W[K,N] f32 -> Wt[N,K] bf16
// ---------------------------------------------------------------------------
__global__ __launch_bounds__(256) void wtrans_kernel(const float* __restrict__ W,
                                                     unsigned short* __restrict__ Wt,
                                                     int K, int N) {
  int idx = blockIdx.x * 256 + threadIdx.x;
  if (idx < K * N) {
    int k = idx / N, n = idx - k * N;
    Wt[(size_t)n * K + k] = f32_to_bf16(W[idx]);
  }
}

// ---------------------------------------------------------------------------
// Kernel 3: generic WMMA GEMM. C[M,N] = A[M,K]@Bt' (+ A2@B2t') + bias(+bias2),
// optional exact-GELU, f32 and/or bf16 output, optional transposed bf16 store.
// One wave computes a 16x64 tile. K is processed in 64-wide chunks: all 10
// fragment loads (20x global_load_b128) are fenced ahead of the 8 WMMAs, so
// waits on the matrix pipe are partial-loadcnt, not full drains.
// Requires K % 64 == 0 (true for every call here).
// ---------------------------------------------------------------------------
__device__ __forceinline__ void gemm_k_chunk(v8f acc[4], const unsigned short* A,
                                             int lda, const unsigned short* Bt,
                                             int ldb, int row0, int col0, int k0,
                                             int lane) {
  Frag16 a0, a1, b0[4], b1[4];
  load_a_frag(a0, A, lda, row0, k0, lane);
  load_a_frag(a1, A, lda, row0, k0 + 32, lane);
#pragma unroll
  for (int j = 0; j < 4; ++j) load_b_frag(b0[j], Bt, ldb, col0 + 16 * j, k0, lane);
#pragma unroll
  for (int j = 0; j < 4; ++j) load_b_frag(b1[j], Bt, ldb, col0 + 16 * j, k0 + 32, lane);
  SCHED_FENCE();  // keep all loads ahead of the WMMA batch
#pragma unroll
  for (int j = 0; j < 4; ++j) acc[j] = wmma_bf16(a0, b0[j], acc[j]);
#pragma unroll
  for (int j = 0; j < 4; ++j) acc[j] = wmma_bf16(a1, b1[j], acc[j]);
}

__global__ __launch_bounds__(256) void gemm_wmma_kernel(
    const unsigned short* __restrict__ A, int lda,
    const unsigned short* __restrict__ Bt, int ldb, int K,
    const unsigned short* __restrict__ A2, int lda2,
    const unsigned short* __restrict__ B2t, int ldb2, int K2,
    const float* __restrict__ bias, const float* __restrict__ bias2,
    float* __restrict__ Cf, unsigned short* __restrict__ Cb, int ldc,
    int M, int N, int act, int tout) {
  int lane = threadIdx.x & 31;
  int wave = threadIdx.x >> 5;
  int tilesN = N >> 6;
  int tile = blockIdx.x * 8 + wave;
  if (tile >= (M >> 4) * tilesN) return;
  int row0 = (tile / tilesN) << 4;
  int col0 = (tile % tilesN) << 6;

  v8f acc[4];
#pragma unroll
  for (int j = 0; j < 4; ++j) acc[j] = vzero();

  for (int k0 = 0; k0 < K; k0 += 64)
    gemm_k_chunk(acc, A, lda, Bt, ldb, row0, col0, k0, lane);
  if (A2 != nullptr) {
    for (int k0 = 0; k0 < K2; k0 += 64)
      gemm_k_chunk(acc, A2, lda2, B2t, ldb2, row0, col0, k0, lane);
  }

  int mbase = (lane >> 4) << 3;
  int n = lane & 15;
#pragma unroll
  for (int j = 0; j < 4; ++j) {
#pragma unroll
    for (int vv = 0; vv < 8; ++vv) {
      int row = row0 + mbase + vv;
      int col = col0 + 16 * j + n;
      float x = acc[j][vv];
      if (bias) x += bias[col];
      if (bias2) x += bias2[col];
      if (act) x = gelu_exact(x);
      if (Cf) Cf[(size_t)row * ldc + col] = x;
      if (Cb) {
        size_t idx = tout ? ((size_t)col * ldc + row) : ((size_t)row * ldc + col);
        Cb[idx] = f32_to_bf16(x);
      }
    }
  }
}

// ---------------------------------------------------------------------------
// Kernel 4: fused attention. One block = (head h, 16 query rows).
// 512 threads = 16 waves, 268KB LDS for the full 16x4096 score strip.
//   phase1: S = Q Kh^T /8 + mask  (WMMA, NT adjacency reads)
//   phase2: softmax per row (float4 LDS passes, wave32 shuffle reductions),
//           stream normalized rows to d_out with nontemporal b128 stores
//   phase3: attended = P @ Vh     (WMMA on bf16 P from LDS, ds_add reduce)
// ---------------------------------------------------------------------------
#define SMEM_SCORES (16 * 4096 * 4)
#define SMEM_ACC (16 * 64 * 4)
#define SMEM_INV 64
#define SMEM_Q (16 * 64 * 2)
#define SMEM_TOTAL (SMEM_SCORES + SMEM_ACC + SMEM_INV + SMEM_Q)

__global__ __launch_bounds__(512) void attn_kernel(
    const unsigned short* __restrict__ Q16, const unsigned short* __restrict__ K16,
    const unsigned short* __restrict__ Vt16, const float* __restrict__ adj,
    const float* __restrict__ swp, float* __restrict__ attn_out,
    unsigned short* __restrict__ att16) {
  extern __shared__ char smem[];
  float* sS = (float*)smem;
  float* sAcc = (float*)(smem + SMEM_SCORES);
  float* sInv = (float*)(smem + SMEM_SCORES + SMEM_ACC);
  unsigned short* sQ = (unsigned short*)(smem + SMEM_SCORES + SMEM_ACC + SMEM_INV);

  const int h = blockIdx.x & 3;
  const int r0 = (blockIdx.x >> 2) << 4;
  const int lane = threadIdx.x & 31;
  const int wave = threadIdx.x >> 5;
  const float maskval = swp[0] * -1e9f;

  // Stage the 16x64 bf16 Q strip into LDS via the CDNA5 async DMA path
  // (tracked by ASYNCcnt), 16 bytes per lane from 4 waves.
  if (threadIdx.x < 128) {
    int rr = threadIdx.x >> 3;
    int cc = (threadIdx.x & 7) << 3;
    const unsigned short* gp = Q16 + (size_t)(r0 + rr) * 256 + h * 64 + cc;
    unsigned lds_off = (unsigned)(size_t)(sQ + rr * 64 + cc);
    asm volatile("global_load_async_to_lds_b128 %0, %1, off"
                 :: "v"(lds_off), "v"(gp)
                 : "memory");
  }
  asm volatile("s_wait_asynccnt 0" ::: "memory");
  __syncthreads();

  // Q A-fragments (k 0..31 and 32..63) built once from LDS, reused for all
  // 16 column tiles handled by this wave.
  Frag16 aq0, aq1;
  {
    const unsigned short* p = sQ + (lane & 15) * 64 + ((lane >> 4) << 3);
    aq0.q[0] = *(const uv4*)p;
    aq0.q[1] = *(const uv4*)(p + 16);
    aq1.q[0] = *(const uv4*)(p + 32);
    aq1.q[1] = *(const uv4*)(p + 48);
  }

  const int mbase = (lane >> 4) << 3;
  const int nn = lane & 15;

  // ---- phase 1: masked scores into LDS ----
  for (int ct = wave; ct < 256; ct += 16) {
    int c0 = ct << 4;
    Frag16 b0, b1;
    load_b_frag(b0, K16 + h * 64, 256, c0, 0, lane);
    load_b_frag(b1, K16 + h * 64, 256, c0, 32, lane);
    SCHED_FENCE();
    v8f acc = vzero();
    acc = wmma_bf16(aq0, b0, acc);
    acc = wmma_bf16(aq1, b1, acc);
#pragma unroll
    for (int vv = 0; vv < 8; ++vv) {
      int m = mbase + vv;
      int c = c0 + nn;
      float s = acc[vv] * 0.125f;  // 1/sqrt(64)
      float av = __builtin_nontemporal_load(adj + (size_t)(r0 + m) * 4096 + c);
      if (av == 0.0f) s += maskval;
      sS[m * 4096 + c] = s;
    }
  }
  __syncthreads();

  // ---- phase 2: softmax, one wave per row, float4 LDS passes;
  //      normalized row streamed out with nontemporal b128 stores ----
  {
    float* rowp = sS + wave * 4096;
    float mx = -3.4e38f;
    for (int c = lane * 4; c < 4096; c += 128) {
      fv4 x = *(const fv4*)(rowp + c);
      mx = fmaxf(fmaxf(fmaxf(mx, x[0]), fmaxf(x[1], x[2])), x[3]);
    }
#pragma unroll
    for (int o = 16; o > 0; o >>= 1) mx = fmaxf(mx, __shfl_xor(mx, o, 32));
    float sum = 0.0f;
    for (int c = lane * 4; c < 4096; c += 128) {
      fv4 x = *(const fv4*)(rowp + c);
#pragma unroll
      for (int e = 0; e < 4; ++e) {
        x[e] = __expf(x[e] - mx);
        sum += x[e];
      }
      *(fv4*)(rowp + c) = x;
    }
#pragma unroll
    for (int o = 16; o > 0; o >>= 1) sum += __shfl_xor(sum, o, 32);
    float inv = 1.0f / sum;
    float* op = attn_out + ((size_t)h * 4096 + (r0 + wave)) * 4096;
    for (int c = lane * 4; c < 4096; c += 128) {
      fv4 x = *(const fv4*)(rowp + c);
#pragma unroll
      for (int e = 0; e < 4; ++e) x[e] *= inv;
      __builtin_nontemporal_store(x, (fv4*)(op + c));
    }
    if (lane == 0) sInv[wave] = inv;
  }
  __syncthreads();

  for (int i = threadIdx.x; i < 16 * 64; i += 512) sAcc[i] = 0.0f;
  __syncthreads();

  // ---- phase 3: attended = P @ V_h, K split across waves ----
  {
    v8f acc[4];
#pragma unroll
    for (int j = 0; j < 4; ++j) acc[j] = vzero();
    const int mm = lane & 15;
    const int kb = (lane >> 4) << 3;
    const float invm = sInv[mm];
    const float* srow = sS + mm * 4096;
    const unsigned short* vbase = Vt16 + (size_t)(h * 64) * 4096;
    for (int ks = wave; ks < 128; ks += 16) {
      int k0 = ks << 5;
      int base = k0 + kb;
      Frag16 b[4];
#pragma unroll
      for (int j = 0; j < 4; ++j) load_b_frag(b[j], vbase, 4096, 16 * j, k0, lane);
      fv4 x0 = *(const fv4*)(srow + base);
      fv4 x1 = *(const fv4*)(srow + base + 4);
      fv4 x2 = *(const fv4*)(srow + base + 16);
      fv4 x3 = *(const fv4*)(srow + base + 20);
      Frag16 a;
#pragma unroll
      for (int e = 0; e < 4; ++e) {
        a.u[e]      = f32_to_bf16(x0[e] * invm);
        a.u[4 + e]  = f32_to_bf16(x1[e] * invm);
        a.u[8 + e]  = f32_to_bf16(x2[e] * invm);
        a.u[12 + e] = f32_to_bf16(x3[e] * invm);
      }
      SCHED_FENCE();
#pragma unroll
      for (int j = 0; j < 4; ++j) acc[j] = wmma_bf16(a, b[j], acc[j]);
    }
#pragma unroll
    for (int j = 0; j < 4; ++j)
#pragma unroll
      for (int vv = 0; vv < 8; ++vv)
        atomicAdd(&sAcc[(mbase + vv) * 64 + 16 * j + nn], acc[j][vv]);
  }
  __syncthreads();

  for (int i = threadIdx.x; i < 16 * 64; i += 512) {
    int m = i >> 6, d = i & 63;
    att16[(size_t)(r0 + m) * 256 + h * 64 + d] = f32_to_bf16(sAcc[i]);
  }
}

// ---------------------------------------------------------------------------
// Kernel 5: layernorm over D=256, one wave per row (8 rows / block).
// Optional elementwise add before normalization; f32 and/or bf16 outputs.
// ---------------------------------------------------------------------------
__global__ __launch_bounds__(256) void ln_kernel(
    const float* __restrict__ X, const float* __restrict__ Xadd,
    const float* __restrict__ g, const float* __restrict__ b,
    float* __restrict__ outF, unsigned short* __restrict__ outB) {
  int lane = threadIdx.x & 31;
  int wave = threadIdx.x >> 5;
  size_t row = (size_t)blockIdx.x * 8 + wave;
  const float* xp = X + row * 256;
  float v[8];
  float s = 0.0f;
#pragma unroll
  for (int i = 0; i < 8; ++i) {
    int c = lane + 32 * i;
    float t = xp[c];
    if (Xadd) t += Xadd[row * 256 + c];
    v[i] = t;
    s += t;
  }
#pragma unroll
  for (int o = 16; o > 0; o >>= 1) s += __shfl_xor(s, o, 32);
  float mean = s * (1.0f / 256.0f);
  float q = 0.0f;
#pragma unroll
  for (int i = 0; i < 8; ++i) {
    float d = v[i] - mean;
    q += d * d;
  }
#pragma unroll
  for (int o = 16; o > 0; o >>= 1) q += __shfl_xor(q, o, 32);
  float rstd = rsqrtf(q * (1.0f / 256.0f) + 1e-5f);
#pragma unroll
  for (int i = 0; i < 8; ++i) {
    int c = lane + 32 * i;
    float o_ = (v[i] - mean) * rstd * g[c] + b[c];
    if (outF) outF[row * 256 + c] = o_;
    if (outB) outB[row * 256 + c] = f32_to_bf16(o_);
  }
}

// ---------------------------------------------------------------------------
extern "C" void kernel_launch(void* const* d_in, const int* in_sizes, int n_in,
                              void* d_out, int out_size, void* d_ws, size_t ws_size,
                              hipStream_t stream) {
  (void)in_sizes; (void)n_in; (void)out_size; (void)ws_size;
  const float* Z = (const float*)d_in[0];
  const float* Zg = (const float*)d_in[1];
  const float* Zc = (const float*)d_in[2];
  const float* adj = (const float*)d_in[3];
  const float* Wq = (const float*)d_in[4];
  const float* bq = (const float*)d_in[5];
  const float* Wk = (const float*)d_in[6];
  const float* bk = (const float*)d_in[7];
  const float* Wv = (const float*)d_in[8];
  const float* bv = (const float*)d_in[9];
  const float* Wo = (const float*)d_in[10];
  const float* bo = (const float*)d_in[11];
  const float* Wr = (const float*)d_in[12];
  const float* br = (const float*)d_in[13];
  const float* ln1g = (const float*)d_in[14];
  const float* ln1b = (const float*)d_in[15];
  const float* ln2g = (const float*)d_in[16];
  const float* ln2b = (const float*)d_in[17];
  const float* f1w = (const float*)d_in[18];
  const float* f1b = (const float*)d_in[19];
  const float* f2w = (const float*)d_in[20];
  const float* f2b = (const float*)d_in[21];
  const float* f3w = (const float*)d_in[22];
  const float* f3b = (const float*)d_in[23];
  const float* sw = (const float*)d_in[24];

  // ---- workspace bump allocator ----
  char* p = (char*)d_ws;
  auto alloc = [&](size_t bytes) -> char* {
    char* r = p;
    p += (bytes + 255) & ~(size_t)255;
    return r;
  };
  const int N = NTOK;
  unsigned short* Zc16 = (unsigned short*)alloc((size_t)N * 128 * 2);
  unsigned short* Zg16 = (unsigned short*)alloc((size_t)N * 128 * 2);
  unsigned short* Z16 = (unsigned short*)alloc((size_t)N * 256 * 2);
  unsigned short* WqT = (unsigned short*)alloc(256 * 128 * 2);
  unsigned short* WkT = (unsigned short*)alloc(256 * 128 * 2);
  unsigned short* WvT = (unsigned short*)alloc(256 * 128 * 2);
  unsigned short* WoT = (unsigned short*)alloc(256 * 256 * 2);
  unsigned short* WrT = (unsigned short*)alloc(256 * 256 * 2);
  unsigned short* F1T = (unsigned short*)alloc(1024 * 256 * 2);
  unsigned short* F2T = (unsigned short*)alloc(512 * 1024 * 2);
  unsigned short* F3T = (unsigned short*)alloc(256 * 512 * 2);
  unsigned short* Q16 = (unsigned short*)alloc((size_t)N * 256 * 2);
  unsigned short* K16 = (unsigned short*)alloc((size_t)N * 256 * 2);
  unsigned short* Vt16 = (unsigned short*)alloc((size_t)256 * N * 2);
  unsigned short* att16 = (unsigned short*)alloc((size_t)N * 256 * 2);
  float* pre1 = (float*)alloc((size_t)N * 256 * 4);
  float* y1 = (float*)alloc((size_t)N * 256 * 4);
  unsigned short* y1_16 = (unsigned short*)alloc((size_t)N * 256 * 2);
  unsigned short* h1_16 = (unsigned short*)alloc((size_t)N * 1024 * 2);
  unsigned short* h2_16 = (unsigned short*)alloc((size_t)N * 512 * 2);
  float* ffn = (float*)alloc((size_t)N * 256 * 4);

  float* outF = (float*)d_out;                      // [4096,256]
  float* attnF = (float*)d_out + (size_t)N * 256;   // [4,4096,4096]

  // ---- 1. activations -> bf16 (+ sinusoidal PE concat) ----
  prep_kernel<<<N, 256, 0, stream>>>(Z, Zg, Zc, Z16, Zg16, Zc16);

  // ---- 2. weights -> transposed bf16 ----
  auto wt = [&](const float* W, unsigned short* Wt, int K, int Nn) {
    wtrans_kernel<<<(K * Nn + 255) / 256, 256, 0, stream>>>(W, Wt, K, Nn);
  };
  wt(Wq, WqT, 128, 256);
  wt(Wk, WkT, 128, 256);
  wt(Wv, WvT, 128, 256);
  wt(Wo, WoT, 256, 256);
  wt(Wr, WrT, 256, 256);
  wt(f1w, F1T, 256, 1024);
  wt(f2w, F2T, 1024, 512);
  wt(f3w, F3T, 512, 256);

  auto gemm = [&](const unsigned short* A, int lda, const unsigned short* Bt,
                  int ldb, int K, const unsigned short* A2, int lda2,
                  const unsigned short* B2t, int ldb2, int K2, const float* bias,
                  const float* bias2, float* Cf, unsigned short* Cb, int ldc,
                  int M, int Nn, int act, int tout) {
    int tiles = (M / 16) * (Nn / 64);
    gemm_wmma_kernel<<<(tiles + 7) / 8, 256, 0, stream>>>(
        A, lda, Bt, ldb, K, A2, lda2, B2t, ldb2, K2, bias, bias2, Cf, Cb, ldc,
        M, Nn, act, tout);
  };

  // ---- 3. QKV projections (V stored transposed per head: Vt[256][4096]) ----
  gemm(Zc16, 128, WqT, 128, 128, nullptr, 0, nullptr, 0, 0, bq, nullptr,
       nullptr, Q16, 256, N, 256, 0, 0);
  gemm(Zg16, 128, WkT, 128, 128, nullptr, 0, nullptr, 0, 0, bk, nullptr,
       nullptr, K16, 256, N, 256, 0, 0);
  gemm(Zg16, 128, WvT, 128, 128, nullptr, 0, nullptr, 0, 0, bv, nullptr,
       nullptr, Vt16, 4096, N, 256, 0, 1);

  // ---- 4. fused attention (writes attn to d_out, attended bf16 to ws) ----
  attn_kernel<<<(N / 16) * 4, 512, SMEM_TOTAL, stream>>>(Q16, K16, Vt16, adj, sw,
                                                         attnF, att16);

  // ---- 5. out-projection + residual projection fused (two K blocks) ----
  gemm(att16, 256, WoT, 256, 256, Z16, 256, WrT, 256, 256, bo, br, pre1,
       nullptr, 256, N, 256, 0, 0);

  // ---- 6. LN1 ----
  ln_kernel<<<N / 8, 256, 0, stream>>>(pre1, nullptr, ln1g, ln1b, y1, y1_16);

  // ---- 7. FFN (GELU fused into GEMM epilogues) ----
  gemm(y1_16, 256, F1T, 256, 256, nullptr, 0, nullptr, 0, 0, f1b, nullptr,
       nullptr, h1_16, 1024, N, 1024, 1, 0);
  gemm(h1_16, 1024, F2T, 1024, 1024, nullptr, 0, nullptr, 0, 0, f2b, nullptr,
       nullptr, h2_16, 512, N, 512, 1, 0);
  gemm(h2_16, 512, F3T, 512, 512, nullptr, 0, nullptr, 0, 0, f3b, nullptr, ffn,
       nullptr, 256, N, 256, 0, 0);

  // ---- 8. LN2 -> final output ----
  ln_kernel<<<N / 8, 256, 0, stream>>>(y1, ffn, ln2g, ln2b, outF, nullptr);
}
